// StreamingLSTM_4389456577211
// MI455X (gfx1250) — matrix operand order repaired and checked by reading, not compiled, not run on hardware
//
#include <hip/hip_runtime.h>
#include <math.h>

// ---------------------------------------------------------------------------
// Streaming LSTM for MI455X (gfx1250, wave32, WMMA).
//   Phase 0: convert W_ih / W_hh fp32 -> fp16 (workspace).
//   Phase 1: gx[t,b,:] = x @ W_ih^T + bias_ih + bias_hh   (v_wmma f16 GEMM)
//   Phase 2: persistent 32-block scan; W_hh slice (256 KB) resident in LDS per
//            WGP, cell state resident in VGPRs, device-wide barrier per step.
// All fragment loads use one per-lane base pointer + compile-time immediate
// offsets; per k-step ALL fragments are loaded into live registers before the
// WMMA group so the scheduler can keep many loads in flight and use graduated
// s_wait_loadcnt/dscnt instead of wait-0 stalls.
// ---------------------------------------------------------------------------

typedef __attribute__((ext_vector_type(16))) _Float16 v16h;
typedef __attribute__((ext_vector_type(8)))  float    v8f;

union HFrag { v16h h; uint4 q[2]; };

#define N_T 512
#define N_B 32
#define N_I 1024
#define N_H 1024
#define N_G 4096

// workspace layout (bytes)
#define OFF_WIH  0u                                // 8 MiB : W_ih fp16 [4096][1024]
#define OFF_WHH  (8u << 20)                        // 8 MiB : W_hh fp16 [4096][1024]
#define OFF_HBUF (16u << 20)                       // 128 KiB : h ping-pong fp16 [2][32][1024]
#define OFF_BAR  ((16u << 20) + (128u << 10))      // 256 B : barrier {cnt, gen}
#define OFF_GX   (17u << 20)                       // 256 MiB : gx fp32 [16384][4096]

__device__ __forceinline__ float sigf(float x) { return 1.0f / (1.0f + __expf(-x)); }

// ---------------- Phase 0: weight conversion fp32 -> fp16 -------------------
__global__ void lstm_cvt_w(const float* __restrict__ wih,
                           const float* __restrict__ whh,
                           _Float16* __restrict__ o_ih,
                           _Float16* __restrict__ o_hh) {
  const int n = N_G * N_I;
  for (int i = blockIdx.x * blockDim.x + threadIdx.x; i < n;
       i += gridDim.x * blockDim.x) {
    o_ih[i] = (_Float16)wih[i];
    o_hh[i] = (_Float16)whh[i];
  }
}

// ---------------- Phase 1: gx = x @ W_ih^T + biases -------------------------
// 4096 blocks x 256 threads (8 waves). Block tile 128x128; wave tile 32x64
// (2 M-tiles x 4 N-tiles). Fragment layouts follow the CDNA5 ISA tables:
//  A (16x32 f16): lane holds row lr, halves k0+hi*8..+7 and k0+16+hi*8..+7
//  B (32x16 f16): lane holds col lr, halves k0+hi*8..+7 and k0+16+hi*8..+7
__global__ __launch_bounds__(256)
void lstm_gx(const _Float16* __restrict__ x,       // [16384][1024]
             const _Float16* __restrict__ wih16,   // [4096][1024]
             const float* __restrict__ bih,
             const float* __restrict__ bhh,
             float* __restrict__ gx) {              // [16384][4096]
  const int lane = threadIdx.x & 31;
  const int wave = threadIdx.x >> 5;   // 0..7 -> 4(M) x 2(N)
  const int wm = wave >> 1;
  const int wn = wave & 1;
  const int m0 = (blockIdx.x & 127) * 128 + wm * 32;
  const int n0 = (blockIdx.x >> 7) * 128 + wn * 64;
  const int lr = lane & 15;
  const int hi = lane >> 4;

  // per-lane base pointers; everything else is an immediate offset
  const _Float16* pA = x + (size_t)(m0 + lr) * N_I + hi * 8;
  const _Float16* pB = wih16 + (size_t)(n0 + lr) * N_I + hi * 8;

  v8f acc[2][4];
#pragma unroll
  for (int i = 0; i < 2; ++i)
#pragma unroll
    for (int j = 0; j < 4; ++j)
#pragma unroll
      for (int r = 0; r < 8; ++r) acc[i][j][r] = 0.0f;

#pragma unroll
  for (int k0 = 0; k0 < N_I; k0 += 32) {
    // load ALL fragments for this k-step first (12 x b128, all live)
    HFrag a[2], bf[4];
#pragma unroll
    for (int i = 0; i < 2; ++i) {
      a[i].q[0] = *(const uint4*)(pA + i * 16 * N_I + k0);
      a[i].q[1] = *(const uint4*)(pA + i * 16 * N_I + k0 + 16);
    }
#pragma unroll
    for (int j = 0; j < 4; ++j) {
      bf[j].q[0] = *(const uint4*)(pB + j * 16 * N_I + k0);
      bf[j].q[1] = *(const uint4*)(pB + j * 16 * N_I + k0 + 16);
    }
    // then the 8-WMMA group
#pragma unroll
    for (int j = 0; j < 4; ++j)
#pragma unroll
      for (int i = 0; i < 2; ++i)
        acc[i][j] = __builtin_amdgcn_wmma_f32_16x16x32_f16(
            false, a[i].h, false, bf[j].h, (short)0, acc[i][j], false, false);
  }

  // epilogue: bias add + fp32 store, base + immediate offsets
  const float* pbi = bih + n0 + lr;
  const float* pbh = bhh + n0 + lr;
  float* pC = gx + (size_t)(m0 + hi * 8) * N_G + n0 + lr;
#pragma unroll
  for (int j = 0; j < 4; ++j) {
    const float bb = pbi[j * 16] + pbh[j * 16];
#pragma unroll
    for (int i = 0; i < 2; ++i)
#pragma unroll
      for (int r = 0; r < 8; ++r)
        pC[(size_t)(i * 16 + r) * N_G + j * 16] = acc[i][j][r] + bb;
  }
}

// ---------------- Phase 2: persistent LSTM scan -----------------------------
// 32 blocks x 128 threads. Block wg owns h columns [wg*32, wg*32+32) -> the
// 128 W_hh rows {g*1024 + wg*32 + c} live in 256 KB of LDS (once). Wave w:
// batch half mt=w&1, column group cg=w>>1; its 4 WMMA accumulators are the
// i/f/g/o gate tiles for its 16x16 (batch x h-col) slot, so the elementwise
// update needs no cross-wave traffic and c stays in VGPRs all 512 steps.
__global__ __launch_bounds__(128)
void lstm_scan(const float* __restrict__ gx,        // [T*B][4096]
               const _Float16* __restrict__ whh16,  // [4096][1024]
               const _Float16* __restrict__ h0,     // [32][1024]
               const _Float16* __restrict__ c0,     // [32][1024]
               _Float16* __restrict__ hbuf,         // [2][32][1024]
               _Float16* __restrict__ y,            // [T][32][1024]
               _Float16* __restrict__ hy,           // [32][1024]
               _Float16* __restrict__ cy,           // [32][1024]
               unsigned* __restrict__ bar) {        // [0]=cnt [1]=gen
  extern __shared__ __align__(16) _Float16 Wsl[];   // [128][1024] = 256 KiB

  const int wg = blockIdx.x;          // 0..31
  const int tid = threadIdx.x;
  const int lane = tid & 31;
  const int wave = tid >> 5;
  const int mt = wave & 1;            // batch half (rows 16*mt ..)
  const int cg = wave >> 1;           // 16-column group within the 32 owned
  const int lr = lane & 15;
  const int hi = lane >> 4;
  const int b0 = mt * 16 + hi * 8;    // first batch row of this lane's slots

  // ---- stage 128x1024 fp16 weight slice into LDS (once) ----
  {
    const uint4* src = (const uint4*)whh16;
    uint4* dst = (uint4*)Wsl;
    for (int idx = tid; idx < 128 * 128; idx += 128) {
      const int rr = idx >> 7;        // slice row 0..127
      const int ch = idx & 127;       // 16-byte chunk within row
      const int g = rr >> 5, c = rr & 31;
      dst[rr * 128 + ch] = src[(size_t)(g * 1024 + wg * 32 + c) * 128 + ch];
    }
  }
  __syncthreads();

  const int colLocal = cg * 16 + lr;       // 0..31
  const int col = wg * 32 + colLocal;      // global h column for this lane

  // per-gate LDS base pointers (hi folded in) -> ds offsets are immediates
  const _Float16* wb[4];
#pragma unroll
  for (int g = 0; g < 4; ++g)
    wb[g] = Wsl + (size_t)(g * 32 + colLocal) * 1024 + hi * 8;

  // cell state lives in VGPRs for the whole scan (fp16-rounded each step)
  v8f cst, hlast;
  {
    const _Float16* pc0 = c0 + (size_t)b0 * N_H + col;
#pragma unroll
    for (int r = 0; r < 8; ++r) {
      cst[r] = (float)pc0[r * N_H];
      hlast[r] = 0.0f;
    }
  }

  const float* pg = gx + (size_t)b0 * N_G + col;        // += B*G per step
  _Float16* py = y + (size_t)b0 * N_H + col;            // += B*H per step
  _Float16* phb0 = hbuf + (size_t)b0 * N_H + col;       // parity 0
  _Float16* phb1 = phb0 + (size_t)N_B * N_H;            // parity 1

  unsigned target = 1;
  for (int t = 0; t < N_T; ++t) {
    const _Float16* hsrc = (t == 0) ? h0 : (hbuf + (size_t)(t & 1) * N_B * N_H);

    // accumulators start at gx (biases already folded in); immediate offsets
    v8f acc[4];
#pragma unroll
    for (int g = 0; g < 4; ++g)
#pragma unroll
      for (int r = 0; r < 8; ++r)
        acc[g][r] = pg[(size_t)r * N_G + g * N_H];

    // pull next step's gx slice toward L2 while this step computes
    if (t + 1 < N_T) {
#pragma unroll
      for (int g = 0; g < 4; ++g)
        __builtin_prefetch(pg + (size_t)N_B * N_G + g * N_H, 0, 1);
    }

    const _Float16* pa = hsrc + (size_t)(mt * 16 + lr) * N_H + hi * 8;
#pragma unroll
    for (int k0 = 0; k0 < N_H; k0 += 32) {
      // all fragments for this k-step live at once (1 global A + 4 LDS B)
      HFrag a, bf[4];
      a.q[0] = *(const uint4*)(pa + k0);
      a.q[1] = *(const uint4*)(pa + k0 + 16);
#pragma unroll
      for (int g = 0; g < 4; ++g) {
        bf[g].q[0] = *(const uint4*)(wb[g] + k0);
        bf[g].q[1] = *(const uint4*)(wb[g] + k0 + 16);
      }
#pragma unroll
      for (int g = 0; g < 4; ++g)
        acc[g] = __builtin_amdgcn_wmma_f32_16x16x32_f16(
            false, a.h, false, bf[g].h, (short)0, acc[g], false, false);
    }

    // elementwise LSTM update for this wave's 16x16 slot
    _Float16* ph = ((t + 1) & 1) ? phb1 : phb0;
#pragma unroll
    for (int r = 0; r < 8; ++r) {
      const float ig = sigf(acc[0][r]);
      const float fg = sigf(acc[1][r]);
      const float gg = tanhf(acc[2][r]);
      const float og = sigf(acc[3][r]);
      float cc = fg * cst[r] + ig * gg;
      const float hh = og * tanhf(cc);             // tanh of unrounded c (ref)
      const _Float16 c16 = (_Float16)cc;
      const _Float16 h16 = (_Float16)hh;
      cst[r] = (float)c16;                         // fp16 round-trip like ref
      hlast[r] = (float)h16;
      ph[r * N_H] = h16;
      py[r * N_H] = h16;
    }
    pg += (size_t)N_B * N_G;
    py += (size_t)N_B * N_H;

    // ---- device-wide barrier over the 32 persistent blocks ----
    __threadfence();
    __syncthreads();
    if (tid == 0) {
      const unsigned old = __hip_atomic_fetch_add(&bar[0], 1u, __ATOMIC_ACQ_REL,
                                                  __HIP_MEMORY_SCOPE_AGENT);
      if (old == 31u) {
        __hip_atomic_store(&bar[0], 0u, __ATOMIC_RELAXED,
                           __HIP_MEMORY_SCOPE_AGENT);
        __hip_atomic_fetch_add(&bar[1], 1u, __ATOMIC_ACQ_REL,
                               __HIP_MEMORY_SCOPE_AGENT);
      } else {
        while (__hip_atomic_load(&bar[1], __ATOMIC_ACQUIRE,
                                 __HIP_MEMORY_SCOPE_AGENT) < target) {
          __builtin_amdgcn_s_sleep(2);
        }
      }
    }
    __syncthreads();
    ++target;
  }

  // final hy / cy
  {
    _Float16* phy = hy + (size_t)b0 * N_H + col;
    _Float16* pcy = cy + (size_t)b0 * N_H + col;
#pragma unroll
    for (int r = 0; r < 8; ++r) {
      phy[r * N_H] = (_Float16)hlast[r];
      pcy[r * N_H] = (_Float16)cst[r];
    }
  }
}

// ---------------------------------------------------------------------------
extern "C" void kernel_launch(void* const* d_in, const int* in_sizes, int n_in,
                              void* d_out, int out_size, void* d_ws,
                              size_t ws_size, hipStream_t stream) {
  (void)in_sizes; (void)n_in; (void)out_size; (void)ws_size;

  const _Float16* x  = (const _Float16*)d_in[0];   // [512][32][1024] fp16
  const _Float16* h0 = (const _Float16*)d_in[1];   // [32][1024] fp16
  const _Float16* c0 = (const _Float16*)d_in[2];   // [32][1024] fp16
  const float* wih   = (const float*)d_in[3];      // [4096][1024] fp32
  const float* whh   = (const float*)d_in[4];      // [4096][1024] fp32
  const float* bih   = (const float*)d_in[5];      // [4096] fp32
  const float* bhh   = (const float*)d_in[6];      // [4096] fp32

  char* ws = (char*)d_ws;
  _Float16* wih16 = (_Float16*)(ws + OFF_WIH);
  _Float16* whh16 = (_Float16*)(ws + OFF_WHH);
  _Float16* hbuf  = (_Float16*)(ws + OFF_HBUF);
  unsigned* bar   = (unsigned*)(ws + OFF_BAR);
  float* gx       = (float*)(ws + OFF_GX);

  _Float16* y  = (_Float16*)d_out;                 // [512][32][1024]
  _Float16* hy = y + (size_t)N_T * N_B * N_H;      // [32][1024]
  _Float16* cy = hy + (size_t)N_B * N_H;           // [32][1024]

  (void)hipMemsetAsync(bar, 0, 256, stream);
  lstm_cvt_w<<<1024, 256, 0, stream>>>(wih, whh, wih16, whh16);
  lstm_gx<<<4096, 256, 0, stream>>>(x, wih16, bih, bhh, gx);
  lstm_scan<<<32, 128, 256 * 1024, stream>>>(gx, whh16, h0, c0, hbuf, y, hy, cy,
                                             bar);
}
